// NaiveAttention_59158879535570
// MI455X (gfx1250) — matrix operand order repaired and checked by reading, not compiled, not run on hardware
//
#include <hip/hip_runtime.h>
#include <hip/hip_bf16.h>

#define D_MODEL 4096
#define N_HEADS 32
#define DK      128
#define MAX_SEQ 4096

typedef __attribute__((ext_vector_type(16))) _Float16 v16h;
typedef __attribute__((ext_vector_type(8)))  float    v8f;

__device__ __forceinline__ float wave_reduce_sum(float v) {
#pragma unroll
  for (int off = 16; off > 0; off >>= 1) v += __shfl_xor(v, off, 32);
  return v;
}
__device__ __forceinline__ float wave_reduce_max(float v) {
#pragma unroll
  for (int off = 16; off > 0; off >>= 1) v = fmaxf(v, __shfl_xor(v, off, 32));
  return v;
}

// ---------------------------------------------------------------------------
// Phase 1: q / k_new / v_new = x @ {W_Q,W_K,W_V}^T.  Wave-per-row GEMV.
// Each lane streams float4 (wave = 512B contiguous per iter) -> coalesced,
// each weight byte read exactly once.  HBM-bound: 192 MB.
// ---------------------------------------------------------------------------
__global__ void qkv_gemv_kernel(const float* __restrict__ x,
                                const float* __restrict__ Wq,
                                const float* __restrict__ Wk,
                                const float* __restrict__ Wv,
                                float* __restrict__ qkv /* 3*D_MODEL */) {
  const int lane = threadIdx.x & 31;
  const int wave = threadIdx.x >> 5;
  const int row  = blockIdx.x * (blockDim.x >> 5) + wave;   // 0..12287
  const int which = row >> 12;
  const int r     = row & (D_MODEL - 1);
  const float* W = (which == 0) ? Wq : (which == 1) ? Wk : Wv;
  const float4* wrow = (const float4*)(W + (size_t)r * D_MODEL);
  const float4* xv   = (const float4*)x;
  float acc = 0.f;
#pragma unroll 4
  for (int it = 0; it < D_MODEL / 128; ++it) {
    const int idx = it * 32 + lane;
    if (it + 2 < D_MODEL / 128) __builtin_prefetch(&wrow[idx + 64], 0, 0);
    const float4 w  = wrow[idx];
    const float4 xx = xv[idx];
    acc += w.x * xx.x + w.y * xx.y + w.z * xx.z + w.w * xx.w;
  }
  acc = wave_reduce_sum(acc);
  if (lane == 0) qkv[row] = acc;
}

// ---------------------------------------------------------------------------
// Phase 2: scores[h][s] = (q[h] . K[s,h]) / sqrt(DK) via v_wmma_f32_16x16x32_f16.
// One wave per (head, 16-position tile).  A = K rows (16 x 32 f16),
// B = q chunk broadcast to all 16 columns, C accumulates in f32.
// Loop d0 over DK in steps of 32 -> 4 WMMA per tile. K slice (64MB) read once.
// A-layout (16-bit A 16x32): lane<16: M=lane, K in {0..7,16..23};
//                            lane>=16: M=lane-16, K in {8..15,24..31}.
// ---------------------------------------------------------------------------
__global__ void attn_scores_kernel(const float* __restrict__ kc,
                                   const float* __restrict__ q,
                                   const float* __restrict__ knew,
                                   const int* __restrict__ seq_idx_p,
                                   const int* __restrict__ cur_pos_p,
                                   float* __restrict__ scores) {
  const int lane = threadIdx.x & 31;
  const int wave = threadIdx.x >> 5;
  const int tile = blockIdx.x * (blockDim.x >> 5) + wave;   // 0..8191
  const int h    = tile >> 8;                               // 256 tiles/head
  const int s0   = (tile & 255) << 4;
  const int seq  = seq_idx_p[0];
  const int cp   = cur_pos_p[0];
  const int half = lane >> 4;
  const int m    = lane & 15;
  const int s    = s0 + m;
  const float* krow = (s == cp)
      ? (knew + h * DK)
      : (kc + (((size_t)seq * MAX_SEQ + s) * N_HEADS + h) * DK);
  const float* qh = q + h * DK;

  v8f c = {};
#pragma unroll
  for (int d0 = 0; d0 < DK; d0 += 32) {
    v16h a, b;
#pragma unroll
    for (int e = 0; e < 16; ++e) {
      const int ka = ((e & 8) << 1) + (e & 7) + (half << 3);  // A: K index
      const int kb = ((e & 8) << 1) + (e & 7) + (half << 3);  // B: K index
      a[e] = (_Float16)krow[d0 + ka];
      b[e] = (_Float16)qh[d0 + kb];      // same for all 16 columns (broadcast)
    }
    c = __builtin_amdgcn_wmma_f32_16x16x32_f16(false, a, false, b,
                                               (short)0, c, false, false);
  }
  // D[m][n] identical over n.  Lane with n==0 in each half holds M = r + 8*half.
  if ((lane & 15) == 0) {
    const float sc = 0.08838834764831845f;   // 1/sqrt(128)
    const int mbase = half << 3;
#pragma unroll
    for (int r = 0; r < 8; ++r)
      scores[(size_t)h * MAX_SEQ + s0 + mbase + r] = c[r] * sc;
  }
}

// ---------------------------------------------------------------------------
// Phase 3: per-head softmax over s in [0, cur_pos]; zero weights beyond so the
// WMMA consumer needs no bounds checks.  Also zeroes the fp32 out accumulator.
// ---------------------------------------------------------------------------
__global__ void softmax_kernel(const float* __restrict__ scores,
                               const int* __restrict__ cur_pos_p,
                               float* __restrict__ attn,
                               float* __restrict__ outbuf) {
  const int h   = blockIdx.x;
  const int tid = threadIdx.x;
  const int len = cur_pos_p[0] + 1;
  if (tid < DK) outbuf[h * DK + tid] = 0.f;
  const float* row  = scores + (size_t)h * MAX_SEQ;
  float*       arow = attn   + (size_t)h * MAX_SEQ;
  __shared__ float red[8];

  float m = -1e30f;
  for (int s = tid; s < len; s += blockDim.x) m = fmaxf(m, row[s]);
  m = wave_reduce_max(m);
  if ((tid & 31) == 0) red[tid >> 5] = m;
  __syncthreads();
  float bm = red[0];
#pragma unroll
  for (int i = 1; i < 8; ++i) bm = fmaxf(bm, red[i]);
  __syncthreads();

  float sum = 0.f;
  for (int s = tid; s < len; s += blockDim.x) sum += __expf(row[s] - bm);
  sum = wave_reduce_sum(sum);
  if ((tid & 31) == 0) red[tid >> 5] = sum;
  __syncthreads();
  float bs = red[0];
#pragma unroll
  for (int i = 1; i < 8; ++i) bs += red[i];
  const float inv = 1.f / bs;

  for (int s = tid; s < MAX_SEQ; s += blockDim.x)
    arow[s] = (s < len) ? __expf(row[s] - bm) * inv : 0.f;
}

// ---------------------------------------------------------------------------
// Phase 4: out[h][d] = sum_s attn[h][s] * V[s,h,d] via WMMA.
// A = attn chunk broadcast across all 16 rows, B = V tile (32 s x 16 d f16),
// D columns give 16 output features (all rows equal).  s split 8 ways across
// waves; partials combined with global_atomic_add_f32.  V (64MB) read once.
// B k order used here (contiguous, same as A): lane<16: N=lane, K=0..15 (2/VGPR);
//                            lane>=16: N=lane-16, K=16..31.
// ---------------------------------------------------------------------------
__global__ void attn_out_kernel(const float* __restrict__ vc,
                                const float* __restrict__ vnew,
                                const float* __restrict__ attn,
                                const int* __restrict__ seq_idx_p,
                                const int* __restrict__ cur_pos_p,
                                float* __restrict__ outbuf) {
  const int lane = threadIdx.x & 31;
  const int wave = threadIdx.x >> 5;
  const int gw   = blockIdx.x * (blockDim.x >> 5) + wave;   // 0..2047
  const int seg  = gw & 7;
  const int pair = gw >> 3;       // 0..255 = (head, d-tile)
  const int h    = pair >> 3;
  const int d0   = (pair & 7) << 4;
  const int seq  = seq_idx_p[0];
  const int cp   = cur_pos_p[0];
  const int half = lane >> 4;
  const int n    = lane & 15;
  const float* arow = attn + (size_t)h * MAX_SEQ;

  v8f c = {};
  const int sbeg = seg * (MAX_SEQ / 8);
  for (int s0 = sbeg; s0 < sbeg + MAX_SEQ / 8; s0 += 32) {
    v16h a, b;
#pragma unroll
    for (int e = 0; e < 16; ++e) {
      const int ka = e + (half << 4);
      a[e] = (_Float16)arow[s0 + ka];          // broadcast over rows
      const int s = s0 + e + (half << 4);      // B K index
      const float* vrow = (s == cp)
          ? (vnew + h * DK)
          : (vc + (((size_t)seq * MAX_SEQ + s) * N_HEADS + h) * DK);
      b[e] = (_Float16)vrow[d0 + n];
    }
    c = __builtin_amdgcn_wmma_f32_16x16x32_f16(false, a, false, b,
                                               (short)0, c, false, false);
  }
  // All M rows equal; lanes 0..15 VGPR0 hold (M=0, N=lane).
  if (lane < 16) atomicAdd(&outbuf[h * DK + d0 + lane], c[0]);
}

// ---------------------------------------------------------------------------
// Phase 5: final = attn_out @ W_O^T.  Wave-per-row GEMV, W_O (64MB) read once.
// ---------------------------------------------------------------------------
__global__ void oproj_kernel(const float* __restrict__ a,
                             const float* __restrict__ Wo,
                             float* __restrict__ out) {
  const int lane = threadIdx.x & 31;
  const int wave = threadIdx.x >> 5;
  const int row  = blockIdx.x * (blockDim.x >> 5) + wave;   // 0..4095
  const float4* wrow = (const float4*)(Wo + (size_t)row * D_MODEL);
  const float4* av   = (const float4*)a;
  float acc = 0.f;
#pragma unroll 4
  for (int it = 0; it < D_MODEL / 128; ++it) {
    const int idx = it * 32 + lane;
    if (it + 2 < D_MODEL / 128) __builtin_prefetch(&wrow[idx + 64], 0, 0);
    const float4 w  = wrow[idx];
    const float4 aa = av[idx];
    acc += w.x * aa.x + w.y * aa.y + w.z * aa.z + w.w * aa.w;
  }
  acc = wave_reduce_sum(acc);
  if (lane == 0) out[row] = acc;
}

extern "C" void kernel_launch(void* const* d_in, const int* in_sizes, int n_in,
                              void* d_out, int out_size, void* d_ws, size_t ws_size,
                              hipStream_t stream) {
  const float* x       = (const float*)d_in[0];
  const int*   seq_idx = (const int*)  d_in[1];
  const int*   cur_pos = (const int*)  d_in[2];
  const float* kc      = (const float*)d_in[3];
  const float* vc      = (const float*)d_in[4];
  const float* Wq      = (const float*)d_in[5];
  const float* Wk      = (const float*)d_in[6];
  const float* Wv      = (const float*)d_in[7];
  const float* Wo      = (const float*)d_in[8];
  float* out = (float*)d_out;

  float* ws     = (float*)d_ws;
  float* q      = ws;                       // 4096
  float* knew   = ws + D_MODEL;             // 4096
  float* vnew   = ws + 2 * D_MODEL;         // 4096
  float* scores = ws + 3 * D_MODEL;         // 32*4096
  float* attn   = scores + (size_t)N_HEADS * MAX_SEQ;   // 32*4096
  float* outbuf = attn   + (size_t)N_HEADS * MAX_SEQ;   // 4096

  // 12288 rows, 8 waves/block
  qkv_gemv_kernel<<<1536, 256, 0, stream>>>(x, Wq, Wk, Wv, ws);
  // 32 heads * 256 tiles = 8192 waves
  attn_scores_kernel<<<1024, 256, 0, stream>>>(kc, q, knew, seq_idx, cur_pos, scores);
  softmax_kernel<<<N_HEADS, 256, 0, stream>>>(scores, cur_pos, attn, outbuf);
  // 32 heads * 8 d-tiles * 8 s-segments = 2048 waves
  attn_out_kernel<<<256, 256, 0, stream>>>(vc, vnew, attn, seq_idx, cur_pos, outbuf);
  // 4096 rows
  oproj_kernel<<<512, 256, 0, stream>>>(outbuf, Wo, out);
}
